// Qwen3Attention_34806414967303
// MI455X (gfx1250) — matrix operand order repaired and checked by reading, not compiled, not run on hardware
//
#include <hip/hip_runtime.h>
#include <hip/hip_bf16.h>

// ---------------------------------------------------------------------------
// Qwen3-style attention for MI455X (gfx1250), bf16 WMMA path.
// Round 4: GEMM re-tiled to 128x128 C-tiles (256 thr / 8 waves, wave=32x64),
// K-step 64, both A and W tiles async-staged to LDS (double buffered, 64KB).
// Attention kernel unchanged (pipelined async K/V staging, transposed V,
// vector ds_load_b128 fragments); final divide -> v_rcp_f32.
// ---------------------------------------------------------------------------

typedef __bf16 bf16_t;
typedef __attribute__((ext_vector_type(16))) __bf16        v16bf;
typedef __attribute__((ext_vector_type(8)))  float         v8f;
typedef __attribute__((ext_vector_type(4)))  unsigned int  v4u;

struct TwoQ { v4u a, b; };
union FragU { v16bf v; TwoQ q; };
union Seg8  { v4u q; bf16_t e[8]; };

static __device__ __forceinline__ v8f wmma_bf16(v16bf a, v16bf b, v8f c) {
  // (neg_a, A, neg_b, B, c_mod, C, reuse_a, reuse_b)
  return __builtin_amdgcn_wmma_f32_16x16x32_bf16(false, a, false, b,
                                                 (short)0, c, false, false);
}

// Async 16B copy global -> LDS (ASYNCcnt tracked).
static __device__ __forceinline__ void async_b128(unsigned ldsOff, const bf16_t* g) {
  const unsigned long long ga = (unsigned long long)(size_t)g;
  asm volatile("global_load_async_to_lds_b128 %0, %1, off"
               :: "v"(ldsOff), "v"(ga) : "memory");
}

// A fragment: 16(M) x 32(K), source row-major, leading dim `ld` (elems).
// lane<16 -> K chunks {0..7,16..23}; lane>=16 -> {8..15,24..31}; row m=lane&15.
static __device__ __forceinline__ v16bf load_fragA(const bf16_t* p, int ld, int lane) {
  const int r  = lane & 15;
  const int k0 = (lane >> 4) << 3;   // 0 or 8
  const bf16_t* row = p + (long)r * ld;
  FragU f;
  f.q.a = *reinterpret_cast<const v4u*>(row + k0);
  f.q.b = *reinterpret_cast<const v4u*>(row + k0 + 16);
  return f.v;
}

// B fragment: 32(K) x 16(N), source row-major N x K (K contiguous per row).
// Column n = lane&15; lane<16 holds K {0..7,16..23}; lane>=16 holds {8..15,24..31}.
static __device__ __forceinline__ v16bf load_fragB(const bf16_t* p, int ld, int lane) {
  const int n  = lane & 15;
  const int k0 = (lane >> 4) << 3;   // 0 or 8
  const bf16_t* row = p + (long)n * ld + k0;
  FragU f;
  f.q.a = *reinterpret_cast<const v4u*>(row);
  f.q.b = *reinterpret_cast<const v4u*>(row + 16);
  return f.v;
}

// ---------------------------------------------------------------------------
__global__ void f32_to_bf16_kernel(const float* __restrict__ in,
                                   bf16_t* __restrict__ out, long n) {
  long i = (long)blockIdx.x * blockDim.x + threadIdx.x;
  const long stride = (long)gridDim.x * blockDim.x;
  for (; i < n; i += stride) out[i] = (bf16_t)in[i];
}

// ---------------------------------------------------------------------------
// C[m,n] = sum_k A[m,k] * W[n,k];  A: MxK bf16 row-major, W: NxK bf16 row-major.
// 256 threads = 8 waves laid out 4(m) x 2(n); wave computes 32x64.
// Block C tile = 128x128.  K-step 64.  A tile (128x64) and W tile (128x64)
// are async-staged into LDS, double buffered (4 x 16KB).
// Requires M%128==0, N%128==0, K%64==0.
template <bool BF16OUT>
__global__ __launch_bounds__(256, 1)
void gemm_xwt_kernel(const bf16_t* __restrict__ A,
                     const bf16_t* __restrict__ W,
                     float* __restrict__ Cf, bf16_t* __restrict__ Cb,
                     int M, int N, int K) {
  __shared__ __align__(16) bf16_t at[2][128 * 64];   // (m, k) 16KB each
  __shared__ __align__(16) bf16_t wt[2][128 * 64];   // (n, k) 16KB each
  const int tid  = threadIdx.x;
  const int lane = tid & 31;
  const int wave = tid >> 5;       // 0..7
  const int wm   = wave >> 1;      // 0..3 -> rows wm*32
  const int wn   = wave & 1;       // 0..1 -> cols wn*64
  const int mblk = blockIdx.x * 128;
  const int nblk = blockIdx.y * 128;

  const unsigned aOff0 = (unsigned)(size_t)(&at[0][0]);
  const unsigned aOff1 = (unsigned)(size_t)(&at[1][0]);
  const unsigned wOff0 = (unsigned)(size_t)(&wt[0][0]);
  const unsigned wOff1 = (unsigned)(size_t)(&wt[1][0]);
  const bf16_t* abase = A + (long)mblk * K;
  const bf16_t* wbase = W + (long)nblk * K;

  // Stage one 128x64 tile: 128 rows x 128B = 1024 segs of 16B, 4 per thread.
  auto stage = [&](const bf16_t* src, int kk, unsigned bufOff) {
#pragma unroll
    for (int i = 0; i < 4; ++i) {
      const int linear = tid + i * 256;     // 0..1023
      const int row = linear >> 3;          // 0..127
      const int seg = linear & 7;           // 16B segment in 128B row
      async_b128(bufOff + (unsigned)(row * 128 + seg * 16),
                 src + (long)row * K + kk + seg * 8);
    }
  };

  v8f acc[2][4] = {};
  stage(abase, 0, aOff0);
  stage(wbase, 0, wOff0);
  asm volatile("s_wait_asynccnt 0" ::: "memory");
  __syncthreads();

  for (int k = 0; k < K; k += 64) {
    const int p = (k >> 6) & 1;
    const bool more = (k + 64) < K;
    if (more) {
      stage(abase, k + 64, p ? aOff0 : aOff1);
      stage(wbase, k + 64, p ? wOff0 : wOff1);
    }
    const bf16_t* atc = &at[p][0];
    const bf16_t* wtc = &wt[p][0];
#pragma unroll
    for (int kk = 0; kk < 64; kk += 32) {
      const v16bf af0 = load_fragA(atc + (wm * 32)      * 64 + kk, 64, lane);
      const v16bf af1 = load_fragA(atc + (wm * 32 + 16) * 64 + kk, 64, lane);
#pragma unroll
      for (int t = 0; t < 4; ++t) {
        const v16bf bf = load_fragB(wtc + (wn * 64 + t * 16) * 64 + kk, 64, lane);
        acc[0][t] = wmma_bf16(af0, bf, acc[0][t]);
        acc[1][t] = wmma_bf16(af1, bf, acc[1][t]);
      }
    }
    asm volatile("s_wait_asynccnt 0" ::: "memory");
    __syncthreads();
  }

  const int ncol  = lane & 15;
  const int rbase = (lane >> 4) * 8;
#pragma unroll
  for (int s = 0; s < 2; ++s) {
#pragma unroll
    for (int t = 0; t < 4; ++t) {
#pragma unroll
      for (int v = 0; v < 8; ++v) {
        const long m = mblk + wm * 32 + s * 16 + rbase + v;
        const long n = nblk + wn * 64 + t * 16 + ncol;
        if (BF16OUT) Cb[m * (long)N + n] = (bf16_t)acc[s][t][v];
        else         Cf[m * (long)N + n] = acc[s][t][v];
      }
    }
  }
}

// ---------------------------------------------------------------------------
// RoPE with fused output scale (1/sqrt(128) for Q, 1.0 for K).
__global__ void rope_kernel(const bf16_t* __restrict__ in,
                            bf16_t* __restrict__ out,
                            const int* __restrict__ pos,
                            int ncols, long total, float scale) {
  const long i = (long)blockIdx.x * blockDim.x + threadIdx.x;
  if (i >= total) return;
  const int  c   = (int)(i % ncols);
  const long row = i / ncols;          // b*S + s
  const int d = c & 127;
  const int j = d & 63;
  const float p   = (float)pos[row];
  const float ang = p * __powf(1.0e6f, -((float)(2 * j)) * (1.0f / 128.0f));
  float sn, cs;
  __sincosf(ang, &sn, &cs);
  const float x = (float)in[i];
  const float other = (d < 64) ? -(float)in[i + 64] : (float)in[i - 64];
  out[i] = (bf16_t)((x * cs + other * sn) * scale);
}

// ---------------------------------------------------------------------------
// Flash attention (non-causal). Q pre-scaled by 1/sqrt(128).
// Block = 128 thr = 4 waves covering a 64-row q-slab of one (b,h).
// Double-buffered pipeline per 32-key chunk:
//   K tile (32x128, row-major)  -> LDS via global_load_async_to_lds_b128
//   V tile -> LDS *transposed* (d, key) with padded stride 40 so PV
//   B-fragments are two ds_load_b128 per fragment.
#define VT_LD 40   // padded row stride (elems) of transposed V; 80B, 16B-aligned
__global__ __launch_bounds__(128, 1)
void attn_kernel(const bf16_t* __restrict__ Q,
                 const bf16_t* __restrict__ K,
                 const bf16_t* __restrict__ V,
                 bf16_t* __restrict__ O) {
  const int S = 2048, H = 32, D = 128, QLD = 4096, KVLD = 512, GROUPS = 8;
  __shared__ __align__(16) bf16_t kt[2][32 * 128];      // (key, d), 8KB each
  __shared__ __align__(16) bf16_t vt[2][128 * VT_LD];   // (d, key) padded, 10KB each
  __shared__ __align__(16) bf16_t plds[4][16 * 32];     // per-wave P tiles, 4KB

  const int tid  = threadIdx.x;
  const int lane = tid & 31;
  const int wave = tid >> 5;
  const int bpb  = S / 64;
  const int bh   = blockIdx.x / bpb;
  const int qb   = blockIdx.x % bpb;
  const int b    = bh / H;
  const int h    = bh % H;
  const int kh   = h / GROUPS;
  const int q0   = qb * 64 + wave * 16;

  // Cache the 16x128 Q tile as four A fragments (pre-scaled by 1/sqrt(D)).
  const bf16_t* qbase = Q + ((long)(b * S + q0)) * QLD + h * D;
  v16bf qf[4];
#pragma unroll
  for (int c = 0; c < 4; ++c) qf[c] = load_fragA(qbase + c * 32, QLD, lane);

  v8f o[8] = {};
  float mst[8], lst[8];
#pragma unroll
  for (int v = 0; v < 8; ++v) { mst[v] = -1e30f; lst[v] = 0.0f; }

  const int rbase = (lane >> 4) * 8;
  const int ncol  = lane & 15;

  const unsigned ktOff0 = (unsigned)(size_t)(&kt[0][0]);
  const unsigned ktOff1 = (unsigned)(size_t)(&kt[1][0]);
  const bf16_t* kbase = K + ((long)(b * S)) * KVLD + kh * D;
  const bf16_t* vbase = V + ((long)(b * S)) * KVLD + kh * D;

  // K tile async stage: 32 rows x 256B = 512 segs of 16B, 4 per thread.
  auto stage_k = [&](int kcN, unsigned bufOff) {
#pragma unroll
    for (int i = 0; i < 4; ++i) {
      const int linear = tid + i * 128;
      const int row = linear >> 4;
      const int seg = linear & 15;
      async_b128(bufOff + (unsigned)(row * 256 + seg * 16),
                 kbase + (long)(kcN + row) * KVLD + seg * 8);
    }
  };
  // V tile: load 4 x b128 per thread into regs (coalesced).
  auto load_v = [&](int kcN, v4u* vs) {
#pragma unroll
    for (int i = 0; i < 4; ++i) {
      const int linear = tid + i * 128;
      const int row = linear >> 4;
      const int seg = linear & 15;
      vs[i] = *reinterpret_cast<const v4u*>(
          vbase + (long)(kcN + row) * KVLD + seg * 8);
    }
  };
  // Scatter the held V segments transposed into (d, key) LDS layout.
  auto store_vt = [&](bf16_t* vtb, const v4u* vs) {
#pragma unroll
    for (int i = 0; i < 4; ++i) {
      const int linear = tid + i * 128;
      const int row = linear >> 4;            // key 0..31
      const int seg = linear & 15;
      Seg8 u; u.q = vs[i];
#pragma unroll
      for (int j = 0; j < 8; ++j)
        vtb[(seg * 8 + j) * VT_LD + row] = u.e[j];
    }
  };

  // ---- prologue: stage chunk 0 into buffer 0 ----
  {
    v4u vs[4];
    stage_k(0, ktOff0);
    load_v(0, vs);
    store_vt(&vt[0][0], vs);
    asm volatile("s_wait_asynccnt 0" ::: "memory");
    __syncthreads();
  }

  for (int kc = 0; kc < S; kc += 32) {
    const int p = (kc >> 5) & 1;
    const bool more = (kc + 32) < S;
    const bf16_t* ktc = &kt[p][0];
    const bf16_t* vtc = &vt[p][0];

    v4u vs[4];
    if (more) {                       // kick off next chunk's staging
      stage_k(kc + 32, p ? ktOff0 : ktOff1);
      load_v(kc + 32, vs);
    }

    // ---- scores: two 16x16 tiles vs keys kc..kc+31 (B-frags from LDS) -----
    v8f s0 = {}, s1 = {};
#pragma unroll
    for (int c = 0; c < 4; ++c) {
      s0 = wmma_bf16(qf[c], load_fragB(ktc + c * 32,            128, lane), s0);
      s1 = wmma_bf16(qf[c], load_fragB(ktc + 16 * 128 + c * 32, 128, lane), s1);
    }

    // ---- online softmax (fp32 stats, 16-lane reductions) ------------------
    float p0[8], p1[8], csc[8];
#pragma unroll
    for (int v = 0; v < 8; ++v) {
      const float a  = s0[v];
      const float bb = s1[v];
      float loc = fmaxf(a, bb);
#pragma unroll
      for (int off = 1; off < 16; off <<= 1)
        loc = fmaxf(loc, __shfl_xor(loc, off, 32));
      const float mn = fmaxf(mst[v], loc);
      const float e0 = __expf(a - mn);
      const float e1 = __expf(bb - mn);
      float rs = e0 + e1;
#pragma unroll
      for (int off = 1; off < 16; off <<= 1)
        rs += __shfl_xor(rs, off, 32);
      const float cs = __expf(mst[v] - mn);
      lst[v] = lst[v] * cs + rs;
      mst[v] = mn;
      csc[v] = cs;
      p0[v] = e0; p1[v] = e1;
    }
#pragma unroll
    for (int t = 0; t < 8; ++t)
#pragma unroll
      for (int v = 0; v < 8; ++v) o[t][v] *= csc[v];

    // ---- D-layout P -> row-major 16x32 bf16 in LDS, re-read as A fragment -
#pragma unroll
    for (int v = 0; v < 8; ++v) {
      plds[wave][(rbase + v) * 32 + ncol]      = (bf16_t)p0[v];
      plds[wave][(rbase + v) * 32 + 16 + ncol] = (bf16_t)p1[v];
    }
    asm volatile("s_wait_dscnt 0" ::: "memory");  // same-wave DS RAW
    const v16bf pa = load_fragA(&plds[wave][0], 32, lane);

    // transposed V scatter for the next chunk (plain DS stores)
    if (more) store_vt(p ? &vt[0][0] : &vt[1][0], vs);

    // ---- P(16x32) x V(32x128): 8 WMMAs, B-frags = 2x ds_load_b128 each ----
#pragma unroll
    for (int t = 0; t < 8; ++t)
      o[t] = wmma_bf16(pa, load_fragB(vtc + t * 16 * VT_LD, VT_LD, lane), o[t]);

    asm volatile("s_wait_asynccnt 0" ::: "memory");
    __syncthreads();   // next tiles ready; everyone done with current tiles
  }

  // ---- normalize (v_rcp_f32 + mul) + store (B*S, 32*128) ------------------
#pragma unroll
  for (int v = 0; v < 8; ++v) lst[v] = __builtin_amdgcn_rcpf(lst[v]);
#pragma unroll
  for (int t = 0; t < 8; ++t) {
#pragma unroll
    for (int v = 0; v < 8; ++v) {
      const float val = o[t][v] * lst[v];
      const long  row = (long)(b * S + q0 + rbase + v);
      O[row * (long)QLD + h * D + t * 16 + ncol] = (bf16_t)val;
    }
  }
}

// ---------------------------------------------------------------------------
extern "C" void kernel_launch(void* const* d_in, const int* in_sizes, int n_in,
                              void* d_out, int out_size, void* d_ws, size_t ws_size,
                              hipStream_t stream) {
  (void)in_sizes; (void)n_in; (void)out_size; (void)ws_size;
  const int  Bn = 2, S = 2048, HID = 3584, NQ = 4096, NKV = 512;
  const long M  = (long)Bn * S;   // 4096 rows

  const float* hs  = (const float*)d_in[0];
  const int*   pid = (const int*)d_in[1];
  const float* Wq  = (const float*)d_in[2];
  const float* Wk  = (const float*)d_in[3];
  const float* Wv  = (const float*)d_in[4];
  const float* Wo  = (const float*)d_in[5];
  float* out = (float*)d_out;

  // ---- carve workspace (~209 MB, 256B aligned chunks) ----
  size_t off = 0;
  char* base = (char*)d_ws;
  auto carve = [&](size_t bytes) -> char* {
    char* p = base + off;
    off += (bytes + 255) & ~(size_t)255;
    return p;
  };
  bf16_t* Xb  = (bf16_t*)carve((size_t)M * HID * 2);
  bf16_t* Wqb = (bf16_t*)carve((size_t)NQ * HID * 2);
  bf16_t* Wkb = (bf16_t*)carve((size_t)NKV * HID * 2);
  bf16_t* Wvb = (bf16_t*)carve((size_t)NKV * HID * 2);
  bf16_t* Wob = (bf16_t*)carve((size_t)HID * NQ * 2);
  bf16_t* Qr  = (bf16_t*)carve((size_t)M * NQ * 2);
  bf16_t* Kr  = (bf16_t*)carve((size_t)M * NKV * 2);
  bf16_t* Vb  = (bf16_t*)carve((size_t)M * NKV * 2);
  bf16_t* Qp  = (bf16_t*)carve((size_t)M * NQ * 2);
  bf16_t* Kp  = (bf16_t*)carve((size_t)M * NKV * 2);
  bf16_t* Ao  = (bf16_t*)carve((size_t)M * NQ * 2);

  auto cv = [&](const float* src, bf16_t* dst, long n) {
    long blocks = (n + 255) / 256;
    if (blocks > 4096) blocks = 4096;
    f32_to_bf16_kernel<<<(int)blocks, 256, 0, stream>>>(src, dst, n);
  };
  cv(hs, Xb,  M * HID);
  cv(Wq, Wqb, (long)NQ * HID);
  cv(Wk, Wkb, (long)NKV * HID);
  cv(Wv, Wvb, (long)NKV * HID);
  cv(Wo, Wob, (long)HID * NQ);

  const dim3 gblk(256, 1, 1);
  // QKV projections (128x128 tiles)
  gemm_xwt_kernel<true><<<dim3((unsigned)(M / 128), NQ / 128),  gblk, 0, stream>>>(
      Xb, Wqb, nullptr, Qr, (int)M, NQ, HID);
  gemm_xwt_kernel<true><<<dim3((unsigned)(M / 128), NKV / 128), gblk, 0, stream>>>(
      Xb, Wkb, nullptr, Kr, (int)M, NKV, HID);
  gemm_xwt_kernel<true><<<dim3((unsigned)(M / 128), NKV / 128), gblk, 0, stream>>>(
      Xb, Wvb, nullptr, Vb, (int)M, NKV, HID);

  // RoPE on Q (fused 1/sqrt(128)) and K
  {
    const float invs = 0.08838834764831845f;
    long tq = M * NQ;
    rope_kernel<<<(unsigned)((tq + 255) / 256), 256, 0, stream>>>(Qr, Qp, pid, NQ, tq, invs);
    long tk = M * NKV;
    rope_kernel<<<(unsigned)((tk + 255) / 256), 256, 0, stream>>>(Kr, Kp, pid, NKV, tk, 1.0f);
  }

  // Attention: grid = B * H * (S/64) = 2048 blocks
  attn_kernel<<<2048, 128, 0, stream>>>(Qp, Kp, Vb, Ao);

  // Output projection -> f32 d_out
  gemm_xwt_kernel<false><<<dim3((unsigned)(M / 128), HID / 128), gblk, 0, stream>>>(
      Ao, Wob, out, nullptr, (int)M, HID, NQ);
}
